// MVSNet_77893526880277
// MI455X (gfx1250) — matrix operand order repaired and checked by reading, not compiled, not run on hardware
//
#include <hip/hip_runtime.h>

// MVSNet forward for MI455X (gfx1250, wave32, WMMA).
// Conv/deconv MACs run through v_wmma_f32_16x16x32_f16: one wave computes a
// 16(pixel) x 16(cout) tile, K = kernelVolume*Cin padded to 32, f32 accum.
// - B (weights) are pre-packed once per launch into f16 in the exact WMMA
//   per-lane layout -> inner loop loads B as one aligned v16h (2x b128).
// - A (activations): per lane the 16 elements are two runs of 8 consecutive
//   k; for CIN%8==0 a run never crosses a tap boundary, so each run is two
//   float4 (b128) loads sharing one clamped/masked validity test. f32->f16
//   uses v_cvt_pk_f16_f32 (2 elems/op) and 32-bit pairwise cndmask masking.
// - All gathers branchless (clamped always-safe addresses + cndmask), so
//   EXEC is untouched in the K-loop and loads issue as clauses.
// Memory stays f32 (126MB cost volume < 192MB L2; 23.3TB/s HBM => whole net
// is ~tens of us of traffic); f16 only in registers feeding WMMA.

typedef __attribute__((ext_vector_type(16))) _Float16 v16h;
typedef __attribute__((ext_vector_type(2)))  __fp16   v2h;   // pkrtz result type
typedef __attribute__((ext_vector_type(8)))  float    v8f;

#define EPSBN 1e-3f
#define CFEAT 32

__device__ __forceinline__ int clampi(int x, int lo, int hi)
{
  return x < lo ? lo : (x > hi ? hi : x);
}

// ---------------------------------------------------------------------------
// Weight packing: f32 HWIO/DHWIO (K x Cout) -> f16 WMMA B-layout tiles.
// Element order: [nt][kc][lane][e], value = wgt[(kc*32 + (e&7) + (e>=8?16:0) + (lane&16?8:0))
// * Cout + nt*16 + (lane&15)], zero-padded for K/Cout tails.
// ---------------------------------------------------------------------------
__global__ void pack_weights_kernel(const float* __restrict__ wgt, int K,
                                    int Cout, int nkc,
                                    _Float16* __restrict__ out)
{
  int idx = blockIdx.x * blockDim.x + threadIdx.x;
  int ntN = (Cout + 15) >> 4;
  int total = ntN * nkc * 512;
  if (idx >= total) return;
  int e    = idx & 15;
  int lane = (idx >> 4) & 31;
  int blk  = idx >> 9;              // nt*nkc + kc
  int kc   = blk % nkc;
  int nt   = blk / nkc;
  int k    = kc * 32 + (e & 7) + ((e >= 8) ? 16 : 0) + ((lane & 16) ? 8 : 0);
  int col  = nt * 16 + (lane & 15);
  float v  = (k < K && col < Cout) ? wgt[k * Cout + col] : 0.f;
  out[idx] = (_Float16)v;
}

// ---------------------------------------------------------------------------
// WMMA implicit-GEMM 2D conv, SAME padding, fused BN/ReLU/bias/residual.
// A (16x32 f16): lane M = lane&15, half e -> k = e + (e>=8?8:0) + (laneHi?8:0)
// B (32x16 f16): pre-packed (see pack_weights_kernel)
// C/D (16x16 f32): lane N = lane&15, vgpr r -> M = r + (laneHi?8:0)
// ---------------------------------------------------------------------------
template <int CIN, int R>
__global__ void conv2d_wmma_kernel(
    const float* __restrict__ in, int H, int W,
    const _Float16* __restrict__ wpk, int stride,
    const float* __restrict__ bn_g, const float* __restrict__ bn_b,
    const float* __restrict__ bn_m, const float* __restrict__ bn_v,
    const float* __restrict__ bias, const float* __restrict__ residual,
    int do_relu,
    float* __restrict__ out, int Ho, int Wo, int Cout)
{
  const int lane = threadIdx.x & 31;
  const int wid  = (blockIdx.x * blockDim.x + threadIdx.x) >> 5;
  const int npix = Ho * Wo;
  const int ntM = (npix + 15) >> 4;
  const int ntN = (Cout + 15) >> 4;
  if (wid >= ntM * ntN) return;           // wave-uniform: EXEC stays all-1s
  const int mt = wid % ntM;
  const int nt = wid / ntM;

  int pt = (Ho - 1) * stride + R - H; if (pt < 0) pt = 0;
  const int padY = pt >> 1;
  pt = (Wo - 1) * stride + R - W; if (pt < 0) pt = 0;
  const int padX = pt >> 1;

  constexpr int K   = R * R * CIN;
  constexpr int NKC = (K + 31) / 32;
  constexpr bool VEC = (CIN % 8 == 0);
  const int mpix = mt * 16 + (lane & 15);
  const int ncol = nt * 16 + (lane & 15);
  const int hiA = (lane & 16) ? 8 : 0;
  const bool mok = (mpix < npix);
  const bool nok = (ncol < Cout);

  const int mp = mok ? mpix : 0;          // branchless clamp before division
  const int oy = mp / Wo;
  const int ox = mp - oy * Wo;
  const int ybase = oy * stride - padY;
  const int xbase = ox * stride - padX;
  const int ncc = nok ? ncol : 0;

  const _Float16* wl = wpk + ((nt * NKC) * 32 + lane) * 16;

  v8f acc = {};
  for (int kc = 0; kc < NKC; ++kc) {
    const int k0 = kc * 32;
    const v16h b = *(const v16h*)(wl + kc * 512);
    union { v16h v; v2h h[8]; } au;
    if constexpr (VEC) {
#pragma unroll
      for (int run = 0; run < 2; ++run) {
        const int kstart = k0 + hiA + run * 16;
        const int tap = kstart / CIN;     // constant divisor; run in one tap
        const int ci0 = kstart - tap * CIN;
        const int r   = tap / R;
        const int s   = tap - r * R;
        const int iy  = ybase + r;
        const int ix  = xbase + s;
        const bool ok = mok && (tap < R * R) &&
                        (iy >= 0) && (iy < H) && (ix >= 0) && (ix < W);
        const int iyc = clampi(iy, 0, H - 1);
        const int ixc = clampi(ix, 0, W - 1);
        const float4* bp =
            (const float4*)(in + (iyc * W + ixc) * CIN + ci0);
        const float4 v0 = bp[0];
        const float4 v1 = bp[1];
        const v2h z2 = {};
        const v2h p0 = __builtin_amdgcn_cvt_pkrtz(v0.x, v0.y);
        const v2h p1 = __builtin_amdgcn_cvt_pkrtz(v0.z, v0.w);
        const v2h p2 = __builtin_amdgcn_cvt_pkrtz(v1.x, v1.y);
        const v2h p3 = __builtin_amdgcn_cvt_pkrtz(v1.z, v1.w);
        const int hb = run * 4;
        au.h[hb + 0] = ok ? p0 : z2;      // 32-bit pairwise cndmask
        au.h[hb + 1] = ok ? p1 : z2;
        au.h[hb + 2] = ok ? p2 : z2;
        au.h[hb + 3] = ok ? p3 : z2;
      }
    } else {
#pragma unroll
      for (int e = 0; e < 16; ++e) {
        const int k = k0 + e + ((e >= 8) ? 8 : 0) + hiA;
        const int tap = k / CIN;
        const int ci  = k - tap * CIN;
        const int r   = tap / R;
        const int s   = tap - r * R;
        const int iy  = ybase + r;
        const int ix  = xbase + s;
        const bool ok = mok && (k < K) &&
                        (iy >= 0) && (iy < H) && (ix >= 0) && (ix < W);
        const int iyc = clampi(iy, 0, H - 1);
        const int ixc = clampi(ix, 0, W - 1);
        const float v = in[(iyc * W + ixc) * CIN + ci];
        au.v[e] = (_Float16)(ok ? v : 0.f);
      }
    }
    acc = __builtin_amdgcn_wmma_f32_16x16x32_f16(false, au.v, false, b,
                                                 (short)0, acc, false, false);
  }

  float scale = 1.f, shift = 0.f;
  if (bn_g) {
    float is = rsqrtf(bn_v[ncc] + EPSBN);
    scale = bn_g[ncc] * is;
    shift = bn_b[ncc] - bn_m[ncc] * scale;
  }
  if (bias) shift += bias[ncc];
#pragma unroll
  for (int r = 0; r < 8; ++r) {
    int p = mt * 16 + r + hiA;
    if (p < npix && nok) {
      float y = acc[r] * scale + shift;
      if (do_relu) y = fmaxf(y, 0.f);
      if (residual) y += residual[p * Cout + ncol];
      out[p * Cout + ncol] = y;
    }
  }
}

// ---------------------------------------------------------------------------
// WMMA implicit-GEMM 3D conv / transposed-conv (3x3x3 kernels).
// TRANSPOSE=false: SAME conv with given stride.
// TRANSPOSE=true : conv_transpose stride 2, SAME (jax padding lo=2,hi=1 ->
//                  input tap t = o + k - 2, valid iff t>=0, even, t/2 < in).
// All CIN here are multiples of 8 -> vector A path always.
// ---------------------------------------------------------------------------
template <int CIN, bool TRANSPOSE>
__global__ void conv3d_wmma_kernel(
    const float* __restrict__ in, int Di, int Hi, int Wi,
    const _Float16* __restrict__ wpk, int stride,
    const float* __restrict__ bn_g, const float* __restrict__ bn_b,
    const float* __restrict__ bn_m, const float* __restrict__ bn_v,
    const float* __restrict__ bias, const float* __restrict__ residual,
    int do_relu,
    float* __restrict__ out, int Do, int Ho, int Wo, int Cout)
{
  const int lane = threadIdx.x & 31;
  const int wid  = (blockIdx.x * blockDim.x + threadIdx.x) >> 5;
  const int nvox = Do * Ho * Wo;
  const int ntM = (nvox + 15) >> 4;
  const int ntN = (Cout + 15) >> 4;
  if (wid >= ntM * ntN) return;
  const int mt = wid % ntM;
  const int nt = wid / ntM;

  int padD = 0, padH = 0, padW = 0;
  if (!TRANSPOSE) {
    int t;
    t = (Do - 1) * stride + 3 - Di; padD = (t > 0 ? t : 0) >> 1;
    t = (Ho - 1) * stride + 3 - Hi; padH = (t > 0 ? t : 0) >> 1;
    t = (Wo - 1) * stride + 3 - Wi; padW = (t > 0 ? t : 0) >> 1;
  }

  constexpr int K   = 27 * CIN;
  constexpr int NKC = (K + 31) / 32;
  const int mvox = mt * 16 + (lane & 15);
  const int ncol = nt * 16 + (lane & 15);
  const int hiA = (lane & 16) ? 8 : 0;
  const bool mok = (mvox < nvox);
  const bool nok = (ncol < Cout);

  const int mv = mok ? mvox : 0;
  const int od = mv / (Ho * Wo);
  const int r2v = mv - od * (Ho * Wo);
  const int oh = r2v / Wo;
  const int ow = r2v - oh * Wo;
  const int dbase = TRANSPOSE ? (od - 2) : (od * stride - padD);
  const int hbase = TRANSPOSE ? (oh - 2) : (oh * stride - padH);
  const int wbase = TRANSPOSE ? (ow - 2) : (ow * stride - padW);
  const int ncc = nok ? ncol : 0;

  const _Float16* wl = wpk + ((nt * NKC) * 32 + lane) * 16;

  v8f acc = {};
  for (int kc = 0; kc < NKC; ++kc) {
    const int k0 = kc * 32;
    const v16h b = *(const v16h*)(wl + kc * 512);
    union { v16h v; v2h h[8]; } au;
#pragma unroll
    for (int run = 0; run < 2; ++run) {
      const int kstart = k0 + hiA + run * 16;
      const int tap = kstart / CIN;       // run never crosses a tap (CIN%8==0)
      const int ci0 = kstart - tap * CIN;
      const int kd  = tap / 9;
      const int rr  = tap - kd * 9;
      const int kh  = rr / 3;
      const int kw  = rr - kh * 3;
      bool ok;
      int id, ih, iw;
      if (!TRANSPOSE) {
        id = dbase + kd; ih = hbase + kh; iw = wbase + kw;
        ok = mok && (tap < 27) &&
             (id >= 0) && (id < Di) && (ih >= 0) && (ih < Hi) &&
             (iw >= 0) && (iw < Wi);
      } else {
        const int td = dbase + kd, th = hbase + kh, tw = wbase + kw;
        ok = mok && (tap < 27) &&
             (td >= 0) && (th >= 0) && (tw >= 0) &&
             !(td & 1) && !(th & 1) && !(tw & 1);
        id = td >> 1; ih = th >> 1; iw = tw >> 1;
        ok = ok && (id < Di) && (ih < Hi) && (iw < Wi);
      }
      const int idc = clampi(id, 0, Di - 1);
      const int ihc = clampi(ih, 0, Hi - 1);
      const int iwc = clampi(iw, 0, Wi - 1);
      const float4* bp =
          (const float4*)(in + ((idc * Hi + ihc) * Wi + iwc) * CIN + ci0);
      const float4 v0 = bp[0];
      const float4 v1 = bp[1];
      const v2h z2 = {};
      const v2h p0 = __builtin_amdgcn_cvt_pkrtz(v0.x, v0.y);
      const v2h p1 = __builtin_amdgcn_cvt_pkrtz(v0.z, v0.w);
      const v2h p2 = __builtin_amdgcn_cvt_pkrtz(v1.x, v1.y);
      const v2h p3 = __builtin_amdgcn_cvt_pkrtz(v1.z, v1.w);
      const int hb = run * 4;
      au.h[hb + 0] = ok ? p0 : z2;        // 32-bit pairwise cndmask
      au.h[hb + 1] = ok ? p1 : z2;
      au.h[hb + 2] = ok ? p2 : z2;
      au.h[hb + 3] = ok ? p3 : z2;
    }
    acc = __builtin_amdgcn_wmma_f32_16x16x32_f16(false, au.v, false, b,
                                                 (short)0, acc, false, false);
  }

  float scale = 1.f, shift = 0.f;
  if (bn_g) {
    float is = rsqrtf(bn_v[ncc] + EPSBN);
    scale = bn_g[ncc] * is;
    shift = bn_b[ncc] - bn_m[ncc] * scale;
  }
  if (bias) shift += bias[ncc];
#pragma unroll
  for (int r = 0; r < 8; ++r) {
    int p = mt * 16 + r + hiA;
    if (p < nvox && nok) {
      float y = acc[r] * scale + shift;
      if (do_relu) y = fmaxf(y, 0.f);
      if (residual) y += residual[p * Cout + ncol];
      out[p * Cout + ncol] = y;
    }
  }
}

// ---------------------------------------------------------------------------
// Projection precompute: rot/trans of src_proj @ inv(ref_proj) for src 1,2.
// ---------------------------------------------------------------------------
__device__ inline void invert4(const float* A, float* inv)
{
  float M[4][8];
  for (int i = 0; i < 4; ++i)
    for (int j = 0; j < 4; ++j) {
      M[i][j]     = A[i * 4 + j];
      M[i][4 + j] = (i == j) ? 1.f : 0.f;
    }
  for (int c = 0; c < 4; ++c) {
    int piv = c;
    for (int r = c + 1; r < 4; ++r)
      if (fabsf(M[r][c]) > fabsf(M[piv][c])) piv = r;
    if (piv != c)
      for (int j = 0; j < 8; ++j) { float t = M[c][j]; M[c][j] = M[piv][j]; M[piv][j] = t; }
    float id = 1.f / M[c][c];
    for (int j = 0; j < 8; ++j) M[c][j] *= id;
    for (int r = 0; r < 4; ++r)
      if (r != c) {
        float f = M[r][c];
        for (int j = 0; j < 8; ++j) M[r][j] -= f * M[c][j];
      }
  }
  for (int i = 0; i < 4; ++i)
    for (int j = 0; j < 4; ++j) inv[i * 4 + j] = M[i][4 + j];
}

__global__ void proj_kernel(const float* __restrict__ projs, float* __restrict__ out)
{
  if (threadIdx.x != 0 || blockIdx.x != 0) return;
  float inv[16];
  invert4(projs, inv);                       // ref = proj_matrices[:,0]
  for (int i = 1; i < 3; ++i) {
    const float* P = projs + 16 * i;
    float M[16];
    for (int r = 0; r < 4; ++r)
      for (int c = 0; c < 4; ++c) {
        float s = 0.f;
        for (int k = 0; k < 4; ++k) s += P[r * 4 + k] * inv[k * 4 + c];
        M[r * 4 + c] = s;
      }
    float* o = out + (i - 1) * 12;
    o[0] = M[0]; o[1] = M[1]; o[2]  = M[2];
    o[3] = M[4]; o[4] = M[5]; o[5]  = M[6];
    o[6] = M[8]; o[7] = M[9]; o[8]  = M[10];
    o[9] = M[3]; o[10] = M[7]; o[11] = M[11];
  }
}

// ---------------------------------------------------------------------------
// Homography warp of src features + on-the-fly variance cost volume.
// One thread per (d,y,x); C=32 channels in registers.
// ---------------------------------------------------------------------------
__global__ void warp_variance_kernel(
    const float* __restrict__ f0, const float* __restrict__ f1,
    const float* __restrict__ f2, const float* __restrict__ pw,
    const float* __restrict__ dv, int D, int H, int W,
    float* __restrict__ var)
{
  int idx = blockIdx.x * blockDim.x + threadIdx.x;
  int total = D * H * W;
  if (idx >= total) return;
  int d = idx / (H * W);
  int r = idx - d * (H * W);
  int y = r / W;
  int x = r - y * W;
  float dep = dv[d];

  float vs[CFEAT], vq[CFEAT];
  const float* rf = f0 + r * CFEAT;
#pragma unroll
  for (int c = 0; c < CFEAT; ++c) { float f = rf[c]; vs[c] = f; vq[c] = f * f; }

  for (int srcI = 0; srcI < 2; ++srcI) {
    const float* P = pw + srcI * 12;
    float fx = (float)x, fy = (float)y;
    float rx = P[0] * fx + P[1] * fy + P[2];
    float ry = P[3] * fx + P[4] * fy + P[5];
    float rz = P[6] * fx + P[7] * fy + P[8];
    float pxx = rx * dep + P[9];
    float pyy = ry * dep + P[10];
    float pzz = rz * dep + P[11];
    float z = (fabsf(pzz) < 1e-6f) ? 1e-6f : pzz;
    float px = pxx / z, py = pyy / z;

    float x0f = floorf(px), y0f = floorf(py);
    float wx1 = px - x0f, wy1 = py - y0f;
    float wx0 = 1.f - wx1, wy0 = 1.f - wy1;
    float cw[4] = { wx0 * wy0, wx1 * wy0, wx0 * wy1, wx1 * wy1 };
    float cx[4] = { x0f, x0f + 1.f, x0f, x0f + 1.f };
    float cy[4] = { y0f, y0f, y0f + 1.f, y0f + 1.f };

    const float* F = srcI ? f2 : f1;
    const float* cp[4];
    float cwv[4];
#pragma unroll
    for (int t = 0; t < 4; ++t) {
      bool valid = (cx[t] >= 0.f) && (cx[t] <= (float)(W - 1)) &&
                   (cy[t] >= 0.f) && (cy[t] <= (float)(H - 1));
      int xi = (int)fminf(fmaxf(cx[t], 0.f), (float)(W - 1));
      int yi = (int)fminf(fmaxf(cy[t], 0.f), (float)(H - 1));
      cp[t]  = F + (yi * W + xi) * CFEAT;
      cwv[t] = valid ? cw[t] : 0.f;
    }
#pragma unroll 4
    for (int c = 0; c < CFEAT; ++c) {
      float g = cwv[0] * cp[0][c] + cwv[1] * cp[1][c] +
                cwv[2] * cp[2][c] + cwv[3] * cp[3][c];
      vs[c] += g;
      vq[c] += g * g;
    }
  }
  const float invN = 1.f / 3.f;
  float* o = var + (size_t)idx * CFEAT;
#pragma unroll
  for (int c = 0; c < CFEAT; ++c) {
    float mu = vs[c] * invN;
    o[c] = vq[c] * invN - mu * mu;
  }
}

// ---------------------------------------------------------------------------
// Softmax over depth + expected depth + 4-tap confidence. One thread/pixel.
// ---------------------------------------------------------------------------
__global__ void softmax_depth_conf_kernel(
    const float* __restrict__ cost, const float* __restrict__ dv,
    int D, int HW, float* __restrict__ depth_out, float* __restrict__ conf_out)
{
  int p = blockIdx.x * blockDim.x + threadIdx.x;
  if (p >= HW) return;
  float pr[64];
  float mx = -3.4e38f;
  for (int d = 0; d < D; ++d) {
    float v = cost[d * HW + p];
    pr[d] = v;
    mx = fmaxf(mx, v);
  }
  float s = 0.f;
  for (int d = 0; d < D; ++d) { float e = expf(pr[d] - mx); pr[d] = e; s += e; }
  float inv = 1.f / s;
  float dep = 0.f, idxf = 0.f;
  for (int d = 0; d < D; ++d) {
    float q = pr[d] * inv;
    pr[d] = q;
    dep  += q * dv[d];
    idxf += q * (float)d;
  }
  int idx = (int)idxf;
  if (idx < 0) idx = 0;
  if (idx > D - 1) idx = D - 1;
  float conf = 0.f;
  for (int j = idx - 1; j <= idx + 2; ++j)
    if (j >= 0 && j < D) conf += pr[j];
  depth_out[p] = dep;
  conf_out[p]  = conf;
}

// ---------------------------------------------------------------------------
// Bilinear resize (half-pixel centers) of ref image + concat depth channel.
// ---------------------------------------------------------------------------
__global__ void resize_concat_kernel(
    const float* __restrict__ img, const float* __restrict__ depth,
    float* __restrict__ out, int IH, int IW, int OH, int OW)
{
  int p = blockIdx.x * blockDim.x + threadIdx.x;
  if (p >= OH * OW) return;
  int y = p / OW, x = p - y * OW;
  float sy = ((float)y + 0.5f) * ((float)IH / (float)OH) - 0.5f;
  float sx = ((float)x + 0.5f) * ((float)IW / (float)OW) - 0.5f;
  float y0f = floorf(sy), x0f = floorf(sx);
  float wy1 = sy - y0f, wx1 = sx - x0f;
  int y0 = clampi((int)y0f, 0, IH - 1);
  int y1 = clampi((int)y0f + 1, 0, IH - 1);
  int x0 = clampi((int)x0f, 0, IW - 1);
  int x1 = clampi((int)x0f + 1, 0, IW - 1);
#pragma unroll
  for (int c = 0; c < 3; ++c) {
    float v = (1.f - wy1) * ((1.f - wx1) * img[(y0 * IW + x0) * 3 + c] +
                             wx1 * img[(y0 * IW + x1) * 3 + c]) +
              wy1 * ((1.f - wx1) * img[(y1 * IW + x0) * 3 + c] +
                     wx1 * img[(y1 * IW + x1) * 3 + c]);
    out[p * 4 + c] = v;
  }
  out[p * 4 + 3] = depth[p];
}

// ===========================================================================
// Host side
// ===========================================================================
struct BNP { const float *b, *g, *m, *v, *w; };

static inline BNP getbn(void* const* d_in, int base)
{
  BNP p;
  p.b = (const float*)d_in[base + 0];
  p.g = (const float*)d_in[base + 1];
  p.m = (const float*)d_in[base + 2];
  p.v = (const float*)d_in[base + 3];
  p.w = (const float*)d_in[base + 4];
  return p;
}

struct Packer { hipStream_t st; _Float16* arena; size_t off; };

static _Float16* pack(Packer& P, const float* w, int K, int Cout)
{
  int nkc = (K + 31) / 32;
  int ntN = (Cout + 15) / 16;
  size_t n = (size_t)ntN * nkc * 512;
  _Float16* dst = P.arena + P.off;
  P.off += n;
  int total = (int)n;
  pack_weights_kernel<<<(total + 255) / 256, 256, 0, P.st>>>(w, K, Cout, nkc, dst);
  return dst;
}

static void launch_conv2d(hipStream_t st, const float* in, int H, int W, int Cin,
                          const _Float16* wpk, int R, int stride,
                          const BNP* bn, const float* bias, const float* res,
                          int relu, float* out, int Ho, int Wo, int Cout)
{
  long tiles = (long)((Ho * Wo + 15) / 16) * ((Cout + 15) / 16);
  int blocks = (int)((tiles + 7) / 8);           // 8 waves / 256-thread block
  const float* g = bn ? bn->g : nullptr;
  const float* b = bn ? bn->b : nullptr;
  const float* m = bn ? bn->m : nullptr;
  const float* v = bn ? bn->v : nullptr;
#define LC2(CI, RR)                                                            \
  conv2d_wmma_kernel<CI, RR><<<blocks, 256, 0, st>>>(                          \
      in, H, W, wpk, stride, g, b, m, v, bias, res, relu, out, Ho, Wo, Cout)
  if      (Cin == 3  && R == 3) LC2(3, 3);
  else if (Cin == 4  && R == 3) LC2(4, 3);
  else if (Cin == 8  && R == 3) LC2(8, 3);
  else if (Cin == 8  && R == 5) LC2(8, 5);
  else if (Cin == 16 && R == 3) LC2(16, 3);
  else if (Cin == 16 && R == 5) LC2(16, 5);
  else if (Cin == 32 && R == 3) LC2(32, 3);
#undef LC2
}

static void launch_conv3d(hipStream_t st, const float* in, int Di, int Hi, int Wi,
                          int Cin, const _Float16* wpk, int stride, int transpose,
                          const BNP* bn, const float* bias, const float* res,
                          int relu, float* out, int Do, int Ho, int Wo, int Cout)
{
  long tiles = (long)((Do * Ho * Wo + 15) / 16) * ((Cout + 15) / 16);
  int blocks = (int)((tiles + 7) / 8);
  const float* g = bn ? bn->g : nullptr;
  const float* b = bn ? bn->b : nullptr;
  const float* m = bn ? bn->m : nullptr;
  const float* v = bn ? bn->v : nullptr;
#define LC3(CI, TR)                                                            \
  conv3d_wmma_kernel<CI, TR><<<blocks, 256, 0, st>>>(                          \
      in, Di, Hi, Wi, wpk, stride, g, b, m, v, bias, res, relu,                \
      out, Do, Ho, Wo, Cout)
  if (!transpose) {
    if      (Cin == 8)  LC3(8,  false);
    else if (Cin == 16) LC3(16, false);
    else if (Cin == 32) LC3(32, false);
    else if (Cin == 64) LC3(64, false);
  } else {
    if      (Cin == 16) LC3(16, true);
    else if (Cin == 32) LC3(32, true);
    else if (Cin == 64) LC3(64, true);
  }
#undef LC3
}

extern "C" void kernel_launch(void* const* d_in, const int* in_sizes, int n_in,
                              void* d_out, int out_size, void* d_ws, size_t ws_size,
                              hipStream_t stream)
{
  (void)in_sizes; (void)n_in; (void)out_size; (void)ws_size;

  // --- input layout (JAX pytree order; dicts key-sorted) -------------------
  // 0 imgs, 1 proj_matrices, 2 depth_values, then params:
  // cost: c0..c6 @3..37 (b,g,m,v,w each), prob_b@38, prob_w@39,
  //       t11@40, t7@45, t9@50
  // feat: final_w@55, layers[0..6]@56+5*i
  // refine: final_w@91, layers[0..2]@92+5*i
  const float* imgs  = (const float*)d_in[0];
  const float* projm = (const float*)d_in[1];
  const float* dvals = (const float*)d_in[2];

  const int IH = 512, IW = 640;
  const int FH = 128, FW = 160;      // feature resolution
  const int D = 48;
  const int HW = FH * FW;

  // --- workspace carve-up (floats) ----------------------------------------
  float* ws = (float*)d_ws;
  size_t off = 0;
  float* projws = ws + off; off += 32;
  float* ping   = ws + off; off += (size_t)IH * IW * 8;
  float* pong   = ws + off; off += (size_t)IH * IW * 8;
  float* feats  = ws + off; off += (size_t)3 * HW * 32;
  float* var    = ws + off; off += (size_t)D * HW * 32;   // reused for t11 out
  float* c0b    = ws + off; off += (size_t)D * HW * 8;
  float* t1b    = ws + off; off += (size_t)24 * 64 * 80 * 16;
  float* c2b    = ws + off; off += (size_t)24 * 64 * 80 * 16;
  float* t3b    = ws + off; off += (size_t)12 * 32 * 40 * 32;
  float* c4b    = ws + off; off += (size_t)12 * 32 * 40 * 32;
  float* t5b    = ws + off; off += (size_t)6 * 16 * 20 * 64;
  float* c6b    = ws + off; off += (size_t)6 * 16 * 20 * 64;
  float* costb  = ws + off; off += (size_t)D * HW;
  float* rxb    = ws + off; off += (size_t)HW * 4;
  float* r1b    = ws + off; off += (size_t)HW * 32;
  float* r2b    = ws + off; off += (size_t)HW * 32;
  // f16 packed-weight arena (1M halfs = 2MB, far more than the ~0.4M needed)
  Packer PK { stream, (_Float16*)(ws + off), 0 };
  off += (size_t)1024 * 1024 / 2;

  float* depth_out  = (float*)d_out;            // [0, HW)
  float* refine_out = (float*)d_out + HW;       // [HW, 2HW)
  float* conf_out   = (float*)d_out + 2 * HW;   // [2HW, 3HW)

  // --- pack all weights into WMMA B-layout f16 ----------------------------
  BNP l0 = getbn(d_in, 56), l1 = getbn(d_in, 61), l2 = getbn(d_in, 66),
      l3 = getbn(d_in, 71), l4 = getbn(d_in, 76), l5 = getbn(d_in, 81),
      l6 = getbn(d_in, 86);
  BNP c0 = getbn(d_in, 3),  c1 = getbn(d_in, 8),  c2 = getbn(d_in, 13),
      c3 = getbn(d_in, 18), c4 = getbn(d_in, 23), c5 = getbn(d_in, 28),
      c6 = getbn(d_in, 33);
  BNP t11 = getbn(d_in, 40), t7 = getbn(d_in, 45), t9 = getbn(d_in, 50);
  BNP r0 = getbn(d_in, 92), r1 = getbn(d_in, 97), r2 = getbn(d_in, 102);
  const float* feat_final_w   = (const float*)d_in[55];
  const float* refine_final_w = (const float*)d_in[91];
  const float* prob_b = (const float*)d_in[38];
  const float* prob_w = (const float*)d_in[39];

  _Float16* pl0 = pack(PK, l0.w, 27, 8);
  _Float16* pl1 = pack(PK, l1.w, 72, 8);
  _Float16* pl2 = pack(PK, l2.w, 200, 16);
  _Float16* pl3 = pack(PK, l3.w, 144, 16);
  _Float16* pl4 = pack(PK, l4.w, 144, 16);
  _Float16* pl5 = pack(PK, l5.w, 400, 32);
  _Float16* pl6 = pack(PK, l6.w, 288, 32);
  _Float16* pff = pack(PK, feat_final_w, 288, 32);
  _Float16* pc0 = pack(PK, c0.w, 864, 8);
  _Float16* pc1 = pack(PK, c1.w, 216, 16);
  _Float16* pc2 = pack(PK, c2.w, 432, 16);
  _Float16* pc3 = pack(PK, c3.w, 432, 32);
  _Float16* pc4 = pack(PK, c4.w, 864, 32);
  _Float16* pc5 = pack(PK, c5.w, 864, 64);
  _Float16* pc6 = pack(PK, c6.w, 1728, 64);
  _Float16* pt7 = pack(PK, t7.w, 1728, 32);
  _Float16* pt9 = pack(PK, t9.w, 864, 16);
  _Float16* pt11 = pack(PK, t11.w, 432, 8);
  _Float16* ppr = pack(PK, prob_w, 216, 1);
  _Float16* pr0 = pack(PK, r0.w, 36, 32);
  _Float16* pr1 = pack(PK, r1.w, 288, 32);
  _Float16* pr2 = pack(PK, r2.w, 288, 32);
  _Float16* prf = pack(PK, refine_final_w, 288, 1);

  // --- feature net for each of the 3 views --------------------------------
  for (int i = 0; i < 3; ++i) {
    const float* im = imgs + (size_t)i * IH * IW * 3;
    float* fo = feats + (size_t)i * HW * 32;
    launch_conv2d(stream, im,   512, 640, 3,  pl0, 3, 1, &l0, 0, 0, 1, ping, 512, 640, 8);
    launch_conv2d(stream, ping, 512, 640, 8,  pl1, 3, 1, &l1, 0, 0, 1, pong, 512, 640, 8);
    launch_conv2d(stream, pong, 512, 640, 8,  pl2, 5, 2, &l2, 0, 0, 1, ping, 256, 320, 16);
    launch_conv2d(stream, ping, 256, 320, 16, pl3, 3, 1, &l3, 0, 0, 1, pong, 256, 320, 16);
    launch_conv2d(stream, pong, 256, 320, 16, pl4, 3, 1, &l4, 0, 0, 1, ping, 256, 320, 16);
    launch_conv2d(stream, ping, 256, 320, 16, pl5, 5, 2, &l5, 0, 0, 1, pong, 128, 160, 32);
    launch_conv2d(stream, pong, 128, 160, 32, pl6, 3, 1, &l6, 0, 0, 1, ping, 128, 160, 32);
    launch_conv2d(stream, ping, 128, 160, 32, pff, 3, 1,
                  nullptr, nullptr, nullptr, 0, fo, 128, 160, 32);
  }

  // --- homography warp + variance cost volume -----------------------------
  proj_kernel<<<1, 32, 0, stream>>>(projm, projws);
  {
    int total = D * HW;
    warp_variance_kernel<<<(total + 255) / 256, 256, 0, stream>>>(
        feats, feats + (size_t)HW * 32, feats + (size_t)2 * HW * 32,
        projws, dvals, D, FH, FW, var);
  }

  // --- cost regularization U-Net ------------------------------------------
  launch_conv3d(stream, var, 48, 128, 160, 32, pc0, 1, 0, &c0, 0, 0, 1, c0b, 48, 128, 160, 8);
  launch_conv3d(stream, c0b, 48, 128, 160, 8,  pc1, 2, 0, &c1, 0, 0, 1, t1b, 24, 64, 80, 16);
  launch_conv3d(stream, t1b, 24, 64, 80, 16,   pc2, 1, 0, &c2, 0, 0, 1, c2b, 24, 64, 80, 16);
  launch_conv3d(stream, c2b, 24, 64, 80, 16,   pc3, 2, 0, &c3, 0, 0, 1, t3b, 12, 32, 40, 32);
  launch_conv3d(stream, t3b, 12, 32, 40, 32,   pc4, 1, 0, &c4, 0, 0, 1, c4b, 12, 32, 40, 32);
  launch_conv3d(stream, c4b, 12, 32, 40, 32,   pc5, 2, 0, &c5, 0, 0, 1, t5b, 6, 16, 20, 64);
  launch_conv3d(stream, t5b, 6, 16, 20, 64,    pc6, 1, 0, &c6, 0, 0, 1, c6b, 6, 16, 20, 64);
  // up path: x = skip + relu(bn(deconv(x)))
  launch_conv3d(stream, c6b, 6, 16, 20, 64,  pt7,  2, 1, &t7,  0, c4b, 1, t3b, 12, 32, 40, 32);
  launch_conv3d(stream, t3b, 12, 32, 40, 32, pt9,  2, 1, &t9,  0, c2b, 1, t1b, 24, 64, 80, 16);
  launch_conv3d(stream, t1b, 24, 64, 80, 16, pt11, 2, 1, &t11, 0, c0b, 1, var, 48, 128, 160, 8);
  launch_conv3d(stream, var, 48, 128, 160, 8, ppr, 1, 0,
                nullptr, prob_b, nullptr, 0, costb, 48, 128, 160, 1);

  // --- softmax / depth / confidence ---------------------------------------
  softmax_depth_conf_kernel<<<(HW + 255) / 256, 256, 0, stream>>>(
      costb, dvals, D, HW, depth_out, conf_out);

  // --- refine net ----------------------------------------------------------
  resize_concat_kernel<<<(HW + 255) / 256, 256, 0, stream>>>(
      imgs, depth_out, rxb, IH, IW, FH, FW);
  launch_conv2d(stream, rxb, 128, 160, 4,  pr0, 3, 1, &r0, 0, 0, 1, r1b, 128, 160, 32);
  launch_conv2d(stream, r1b, 128, 160, 32, pr1, 3, 1, &r1, 0, 0, 1, r2b, 128, 160, 32);
  launch_conv2d(stream, r2b, 128, 160, 32, pr2, 3, 1, &r2, 0, 0, 1, r1b, 128, 160, 32);
  launch_conv2d(stream, r1b, 128, 160, 32, prf, 3, 1,
                nullptr, nullptr, depth_out, 0, refine_out, 128, 160, 1);
}
